// CAGroup3DHead_69664369541569
// MI455X (gfx1250) — matrix operand order repaired and checked by reading, not compiled, not run on hardware
//
#include <hip/hip_runtime.h>
#include <hip/hip_bf16.h>

#define NPTS 65536
#define CDIM 128
#define NCLS 10
#define KNBR 27

typedef __attribute__((ext_vector_type(16))) __bf16 v16bf;
typedef __attribute__((ext_vector_type(8)))  __bf16 v8bf;
typedef __attribute__((ext_vector_type(8)))  float  v8f;

union V16U { v16bf v; v8bf h[2]; };

__device__ __forceinline__ float elu1(float x) {
    return x > 0.f ? x : (__expf(x) - 1.f);
}

__device__ __forceinline__ void zero_acc(v8f* acc, int n) {
#pragma unroll
    for (int i = 0; i < 8; ++i) {
        if (i >= n) break;
#pragma unroll
        for (int j = 0; j < 8; ++j) acc[i][j] = 0.f;
    }
}

// One wave computes a 16 x (NT*16) tile: A = 16 rows of sA (row-major bf16,
// stride aStride), B = Wt rows n (Wt is W transposed, row-major, stride 128).
// K = kTiles*32, accumulated into acc (f32). All NT B-fragments are loaded
// into distinct live registers before the WMMA chain so the dscnt waits are
// graduated (load latency overlapped with WMMA) instead of serialized.
template<int NT>
__device__ __forceinline__ void wave_gemm(const __bf16* sA, int aStride,
                                          const __bf16* sWt, int kTiles,
                                          v8f* acc, int lane, int waveRow)
{
    const int m     = lane & 15;
    const int off8  = (lane & 16) ? 8  : 0;   // A: K-half select
    const int off16 = (lane & 16) ? 8  : 0;   // B: K-half select
    const __bf16* aRow = sA + (waveRow + m) * aStride + off8;
    const __bf16* bRow = sWt + m * 128 + off16;

    for (int kt = 0; kt < kTiles; ++kt) {
        const int kb = kt * 32;
        V16U a;
        a.h[0] = *(const v8bf*)(aRow + kb);        // j=0..7  -> K=kb+off8+0..7
        a.h[1] = *(const v8bf*)(aRow + kb + 16);   // j=8..15 -> K=kb+off8+16..23
        V16U b[NT];
#pragma unroll
        for (int nt = 0; nt < NT; ++nt) {
            const __bf16* bp = bRow + nt * 16 * 128 + kb;
            b[nt].h[0] = *(const v8bf*)(bp);       // j=0..7  -> K=kb+off16+0..7
            b[nt].h[1] = *(const v8bf*)(bp + 16);  // j=8..15 -> K=kb+off16+16..23
        }
#pragma unroll
        for (int nt = 0; nt < NT; ++nt)
            acc[nt] = __builtin_amdgcn_wmma_f32_16x16x32_bf16(
                false, a.v, false, b[nt].v, (short)0, acc[nt], false, false);
    }
}

// acc (16x128 per wave) -> affine(g,b) -> ELU -> bf16 into sOut (stride 128)
__device__ __forceinline__ void epilogue_lds(const v8f* acc, __bf16* sOut,
                                             const float* __restrict__ g,
                                             const float* __restrict__ b,
                                             int lane, int waveRow)
{
    const int nlo = lane & 15;
    const int mhi = (lane & 16) ? 8 : 0;
#pragma unroll
    for (int nt = 0; nt < 8; ++nt) {
        const int cc = nt * 16 + nlo;
        const float gg = g[cc], bb = b[cc];
#pragma unroll
        for (int v = 0; v < 8; ++v) {
            const int rr = waveRow + v + mhi;
            sOut[rr * 128 + cc] = (__bf16)elu1(acc[nt][v] * gg + bb);
        }
    }
}

// ---------------- prep: bf16 casts + transposed weights ----------------
__global__ void prep_kernel(const float* __restrict__ feats,
                            const float* __restrict__ fo_w,
                            const float* __restrict__ cls_out_w,
                            const float* __restrict__ up_w,
                            const float* __restrict__ fuse_w,
                            const float* __restrict__ exp_w,
                            const float* __restrict__ ctr_w,
                            const float* __restrict__ reg_w,
                            const float* __restrict__ cls_w,
                            __bf16* __restrict__ feats_bf,
                            __bf16* __restrict__ foT, __bf16* __restrict__ clsT,
                            __bf16* __restrict__ upT, __bf16* __restrict__ fuseT,
                            __bf16* __restrict__ expT, __bf16* __restrict__ headT)
{
    const int stride = gridDim.x * blockDim.x;
    const int t0 = blockIdx.x * blockDim.x + threadIdx.x;
    for (int i = t0; i < NPTS * CDIM; i += stride)
        feats_bf[i] = (__bf16)feats[i];
    for (int i = t0; i < KNBR * 16384; i += stride) {
        int k = i / 16384, r = i % 16384, n = r >> 7, kk = r & 127;
        foT[i] = (__bf16)fo_w[k * 16384 + kk * 128 + n];
    }
    for (int i = t0; i < NCLS * 16384; i += stride) {
        int k = i / 16384, r = i % 16384, n = r >> 7, kk = r & 127;
        clsT[i] = (__bf16)cls_out_w[k * 16384 + kk * 128 + n];
        upT[i]  = (__bf16)up_w[k * 16384 + kk * 128 + n];
        expT[i] = (__bf16)exp_w[k * 16384 + kk * 128 + n];
    }
    for (int i = t0; i < NCLS * 32768; i += stride) {
        int c = i / 32768, r = i % 32768, n = r >> 8, kk = r & 255;
        fuseT[i] = (__bf16)fuse_w[c * 32768 + kk * 128 + n];
    }
    for (int i = t0; i < 32 * 128; i += stride) {
        int n = i >> 7, kk = i & 127;
        float v = 0.f;
        if (n == 0)       v = ctr_w[kk];
        else if (n <= 6)  v = reg_w[kk * 6 + (n - 1)];
        else if (n <= 16) v = cls_w[kk * 10 + (n - 7)];
        headT[i] = (__bf16)v;
    }
}

// ---------------- semantic gating mask ----------------
__global__ void sem_kernel(const float* __restrict__ feats,
                           const float* __restrict__ Wsem,
                           const float* __restrict__ bsem,
                           unsigned* __restrict__ maskbits)
{
    const int row = blockIdx.x * blockDim.x + threadIdx.x;
    if (row >= NPTS) return;
    float acc[NCLS];
#pragma unroll
    for (int c = 0; c < NCLS; ++c) acc[c] = 0.f;
    const float* fr = feats + (size_t)row * CDIM;
    for (int j = 0; j < CDIM; ++j) {
        const float f = fr[j];
#pragma unroll
        for (int c = 0; c < NCLS; ++c) acc[c] += f * Wsem[j * NCLS + c];
    }
    const float thrLogit = -1.7346010553881064f;  // log(0.15/0.85)
    unsigned bits = 0;
#pragma unroll
    for (int c = 0; c < NCLS; ++c)
        if (acc[c] + bsem[c] > thrLogit) bits |= (1u << c);
    maskbits[row] = bits;
}

// ---------------- 27-neighbor sparse conv -> BN -> ELU ----------------
__global__ __launch_bounds__(128) void nconv_kernel(
    const __bf16* __restrict__ feats_bf, const int* __restrict__ nbr,
    const __bf16* __restrict__ foT,
    const float* __restrict__ fo_g, const float* __restrict__ fo_b,
    __bf16* __restrict__ off_bf)
{
    __shared__ __attribute__((aligned(16))) __bf16 sW[128 * 128];
    __shared__ __attribute__((aligned(16))) __bf16 sX[64 * 128];
    const int tid = threadIdx.x, lane = tid & 31, wave = tid >> 5;
    const int waveRow = wave * 16;
    const int rowBase = blockIdx.x * 64;

    v8f acc[8];
    zero_acc(acc, 8);
    for (int k = 0; k < KNBR; ++k) {
        for (int i = tid; i < 64 * 16; i += 128) {
            int row = i >> 4, ch = i & 15;
            int idx = nbr[(size_t)(rowBase + row) * KNBR + k];
            *(uint4*)&sX[row * 128 + ch * 8] =
                *(const uint4*)(feats_bf + (size_t)idx * 128 + ch * 8);
        }
        const uint4* wsrc = (const uint4*)(foT + (size_t)k * 16384);
        for (int i = tid; i < 2048; i += 128) ((uint4*)sW)[i] = wsrc[i];
        __syncthreads();
        wave_gemm<8>(sX, 128, sW, 4, acc, lane, waveRow);
        __syncthreads();
    }
    const int nlo = lane & 15, mhi = (lane & 16) ? 8 : 0;
#pragma unroll
    for (int nt = 0; nt < 8; ++nt) {
        int cc = nt * 16 + nlo;
        float g = fo_g[cc], b = fo_b[cc];
#pragma unroll
        for (int v = 0; v < 8; ++v) {
            int rr = rowBase + waveRow + v + mhi;
            off_bf[(size_t)rr * 128 + cc] = (__bf16)elu1(acc[nt][v] * g + b);
        }
    }
}

// ---------------- per-class fused head chain ----------------
__global__ __launch_bounds__(128) void chain_kernel(
    const __bf16* __restrict__ feats_bf, const __bf16* __restrict__ off_bf,
    const unsigned* __restrict__ maskbits,
    const __bf16* __restrict__ clsT, const __bf16* __restrict__ upT,
    const __bf16* __restrict__ fuseT, const __bf16* __restrict__ expT,
    const __bf16* __restrict__ headT,
    const float* __restrict__ cls_out_g, const float* __restrict__ cls_out_b,
    const float* __restrict__ up_g, const float* __restrict__ up_b,
    const float* __restrict__ fuse_g, const float* __restrict__ fuse_b,
    const float* __restrict__ exp_g, const float* __restrict__ exp_b,
    const float* __restrict__ cls_bias, const float* __restrict__ scales,
    float* __restrict__ out)
{
    __shared__ __attribute__((aligned(16))) __bf16 sW[128 * 128];
    __shared__ __attribute__((aligned(16))) __bf16 sX[64 * 128];
    __shared__ __attribute__((aligned(16))) __bf16 sH[64 * 128];
    __shared__ __attribute__((aligned(16))) __bf16 sU[64 * 128];

    const int tid = threadIdx.x, lane = tid & 31, wave = tid >> 5;
    const int waveRow = wave * 16;
    const int c = blockIdx.y;
    const int r2base = blockIdx.x * 64;

    // stage in masked fuse features (+ cls_out weights)
    {
        uint4 z; z.x = z.y = z.z = z.w = 0u;
        for (int i = tid; i < 64 * 16; i += 128) {
            int row = i >> 4, ch = i & 15;
            int r2 = r2base + row;
            int rmod = (r2 < NPTS) ? r2 : r2 - NPTS;
            const __bf16* src =
                ((r2 < NPTS) ? off_bf : feats_bf) + (size_t)rmod * 128 + ch * 8;
            bool mk = (maskbits[rmod] >> c) & 1u;
            *(uint4*)&sX[row * 128 + ch * 8] = mk ? *(const uint4*)src : z;
        }
        const uint4* wsrc = (const uint4*)(clsT + (size_t)c * 16384);
        for (int i = tid; i < 2048; i += 128) ((uint4*)sW)[i] = wsrc[i];
    }
    __syncthreads();

    v8f acc[8];
    // hc = ELU(BN(x @ cls_out_w))
    zero_acc(acc, 8);
    wave_gemm<8>(sX, 128, sW, 4, acc, lane, waveRow);
    epilogue_lds(acc, sH, cls_out_g + c * 128, cls_out_b + c * 128, lane, waveRow);
    __syncthreads();

    // uc = ELU(BN(hc @ up_w))
    { const uint4* wsrc = (const uint4*)(upT + (size_t)c * 16384);
      for (int i = tid; i < 2048; i += 128) ((uint4*)sW)[i] = wsrc[i]; }
    __syncthreads();
    zero_acc(acc, 8);
    wave_gemm<8>(sH, 128, sW, 4, acc, lane, waveRow);
    epilogue_lds(acc, sU, up_g + c * 128, up_b + c * 128, lane, waveRow);
    __syncthreads();

    // fc = ELU(BN([hc, uc] @ fuse_w)) : K=256 in two halves
    for (int i = tid; i < 2048; i += 128) {
        int n = i >> 4, ch = i & 15;
        *(uint4*)&sW[n * 128 + ch * 8] =
            *(const uint4*)(fuseT + (size_t)c * 32768 + n * 256 + ch * 8);
    }
    __syncthreads();
    zero_acc(acc, 8);
    wave_gemm<8>(sH, 128, sW, 4, acc, lane, waveRow);
    __syncthreads();
    for (int i = tid; i < 2048; i += 128) {
        int n = i >> 4, ch = i & 15;
        *(uint4*)&sW[n * 128 + ch * 8] =
            *(const uint4*)(fuseT + (size_t)c * 32768 + n * 256 + 128 + ch * 8);
    }
    __syncthreads();
    wave_gemm<8>(sU, 128, sW, 4, acc, lane, waveRow);
    epilogue_lds(acc, sX, fuse_g + c * 128, fuse_b + c * 128, lane, waveRow);
    __syncthreads();

    // ec = ELU(BN(fc @ exp_w))
    { const uint4* wsrc = (const uint4*)(expT + (size_t)c * 16384);
      for (int i = tid; i < 2048; i += 128) ((uint4*)sW)[i] = wsrc[i]; }
    __syncthreads();
    zero_acc(acc, 8);
    wave_gemm<8>(sX, 128, sW, 4, acc, lane, waveRow);
    epilogue_lds(acc, sH, exp_g + c * 128, exp_b + c * 128, lane, waveRow);
    __syncthreads();

    // heads: [ctr | exp(scale*reg) | cls+bias] * mask
    { const uint4* wsrc = (const uint4*)headT;
      for (int i = tid; i < 512; i += 128) ((uint4*)sW)[i] = wsrc[i]; }
    __syncthreads();
    v8f hacc[2];
    zero_acc(hacc, 2);
    wave_gemm<2>(sH, 128, sW, 4, hacc, lane, waveRow);

    const float sc = scales[c];
    const int nlo = lane & 15, mhi = (lane & 16) ? 8 : 0;
#pragma unroll
    for (int nt = 0; nt < 2; ++nt) {
        int col = nt * 16 + nlo;
        if (col < 17) {
#pragma unroll
            for (int v = 0; v < 8; ++v) {
                int rr = waveRow + v + mhi;
                int r2 = r2base + rr;
                int rmod = (r2 < NPTS) ? r2 : r2 - NPTS;
                float mk = ((maskbits[rmod] >> c) & 1u) ? 1.f : 0.f;
                float x = hacc[nt][v];
                float val = (col == 0) ? x
                          : (col < 7)  ? __expf(sc * x)
                                       : (x + cls_bias[col - 7]);
                out[((size_t)c * (2 * NPTS) + r2) * 17 + col] = val * mk;
            }
        }
    }
}

extern "C" void kernel_launch(void* const* d_in, const int* in_sizes, int n_in,
                              void* d_out, int out_size, void* d_ws, size_t ws_size,
                              hipStream_t stream)
{
    const float* feats     = (const float*)d_in[1];
    const int*   nbr       = (const int*)d_in[2];
    const float* Wsem      = (const float*)d_in[3];
    const float* bsem      = (const float*)d_in[4];
    const float* fo_w      = (const float*)d_in[12];
    const float* fo_g      = (const float*)d_in[13];
    const float* fo_b      = (const float*)d_in[14];
    const float* cls_out_w = (const float*)d_in[15];
    const float* cls_out_g = (const float*)d_in[16];
    const float* cls_out_b = (const float*)d_in[17];
    const float* up_w      = (const float*)d_in[18];
    const float* up_g      = (const float*)d_in[19];
    const float* up_b      = (const float*)d_in[20];
    const float* fuse_w    = (const float*)d_in[21];
    const float* fuse_g    = (const float*)d_in[22];
    const float* fuse_b    = (const float*)d_in[23];
    const float* exp_w     = (const float*)d_in[24];
    const float* exp_g     = (const float*)d_in[25];
    const float* exp_b     = (const float*)d_in[26];
    const float* ctr_w     = (const float*)d_in[27];
    const float* reg_w     = (const float*)d_in[28];
    const float* cls_w     = (const float*)d_in[29];
    const float* cls_b     = (const float*)d_in[30];
    const float* scales    = (const float*)d_in[31];

    char* ws = (char*)d_ws;
    size_t o = 0;
    __bf16* feats_bf = (__bf16*)(ws + o); o += (size_t)NPTS * CDIM * 2;
    __bf16* off_bf   = (__bf16*)(ws + o); o += (size_t)NPTS * CDIM * 2;
    __bf16* foT      = (__bf16*)(ws + o); o += (size_t)KNBR * 16384 * 2;
    __bf16* clsT     = (__bf16*)(ws + o); o += (size_t)NCLS * 16384 * 2;
    __bf16* upT      = (__bf16*)(ws + o); o += (size_t)NCLS * 16384 * 2;
    __bf16* fuseT    = (__bf16*)(ws + o); o += (size_t)NCLS * 32768 * 2;
    __bf16* expT     = (__bf16*)(ws + o); o += (size_t)NCLS * 16384 * 2;
    __bf16* headT    = (__bf16*)(ws + o); o += (size_t)32 * 128 * 2;
    unsigned* maskb  = (unsigned*)(ws + o); o += (size_t)NPTS * 4;

    prep_kernel<<<dim3(1024), dim3(256), 0, stream>>>(
        feats, fo_w, cls_out_w, up_w, fuse_w, exp_w, ctr_w, reg_w, cls_w,
        feats_bf, foT, clsT, upT, fuseT, expT, headT);
    sem_kernel<<<dim3(NPTS / 256), dim3(256), 0, stream>>>(
        feats, Wsem, bsem, maskb);
    nconv_kernel<<<dim3(NPTS / 64), dim3(128), 0, stream>>>(
        feats_bf, nbr, foT, fo_g, fo_b, off_bf);
    chain_kernel<<<dim3((2 * NPTS) / 64, NCLS), dim3(128), 0, stream>>>(
        feats_bf, off_bf, maskb, clsT, upT, fuseT, expT, headT,
        cls_out_g, cls_out_b, up_g, up_b, fuse_g, fuse_b,
        exp_g, exp_b, cls_b, scales, (float*)d_out);
}
